// CliffordSelfAttention_35966056137145
// MI455X (gfx1250) — compile-verified, matrix operand order checked by reading, not run on hardware
//
#include <hip/hip_runtime.h>
#include <cstdint>

#define B_  4
#define S_  4096
#define WAVES 4

typedef __attribute__((ext_vector_type(16))) _Float16 v16h;
typedef __attribute__((ext_vector_type(8)))  _Float16 v8h;
typedef __attribute__((ext_vector_type(8)))  float    v8f;

#define LOG2E 1.44269504088896340736f

// ---- Cl(3,0) geometric product: e_i * e_j = sign * e_k ---------------------
// Blade order [1,e1,e2,e3,e12,e13,e23,e123] <-> bitmasks {0,1,2,4,3,5,6,7}
// (an involution, so the table inverts itself). Unrolled callers fold to
// immediates.
__device__ __forceinline__ void geo(int i, int j, int& k, float& s) {
    const int mask[8] = {0, 1, 2, 4, 3, 5, 6, 7};
    int a = mask[i], b = mask[j];
    int cnt = __popc((a >> 1) & b) + __popc((a >> 2) & b);
    k = mask[a ^ b];
    s = (cnt & 1) ? -1.0f : 1.0f;
}

// ---- Pass 1: per-token Clifford projections ------------------------------
//   Qh16 [B*S][16] : dims 0..7 = (0.5*log2e)*Q, dims 8..15 = 0  (scores come
//                    out in log2 space -> raw v_exp_f32, no per-score mul;
//                    zeros feed bQ's dead K-slots so score A-slots are d.c.)
//   Kh16 [B*S][16] : dims 0..7 = metric-signed K, dims 8..15 = 0
//   Vt16 [B][16][S]: rows 0..7 = V^T, row 8 = 1.0 (softmax-denominator row:
//                    output row 8 accumulates sum(p) inside the PV WMMA),
//                    rows 9..15 = 0 (loaded as A rows; their output rows unused)
__global__ __launch_bounds__(256) void clifford_qkv_kernel(
    const float* __restrict__ x,
    const float* __restrict__ wq, const float* __restrict__ bq,
    const float* __restrict__ wk, const float* __restrict__ bk,
    const float* __restrict__ wv, const float* __restrict__ bv,
    _Float16* __restrict__ Qh16, _Float16* __restrict__ Kh16,
    _Float16* __restrict__ Vt16)
{
    int t = blockIdx.x * blockDim.x + threadIdx.x;
    if (t >= B_ * S_) return;

    float xr[8];
#pragma unroll
    for (int j = 0; j < 8; ++j) xr[j] = x[t * 8 + j];

    float q[8], k[8], v[8];
#pragma unroll
    for (int n = 0; n < 8; ++n) { q[n] = bq[n]; k[n] = bk[n]; v[n] = bv[n]; }

#pragma unroll
    for (int i = 0; i < 8; ++i) {
        float wqi = wq[i], wki = wk[i], wvi = wv[i];
#pragma unroll
        for (int j = 0; j < 8; ++j) {
            int kk; float s;
            geo(i, j, kk, s);
            float sx = s * xr[j];
            q[kk] += wqi * sx;
            k[kk] += wki * sx;
            v[kk] += wvi * sx;
        }
    }

    int b = t / S_, si = t % S_;
#pragma unroll
    for (int n = 0; n < 8; ++n) {
        Qh16[t * 16 + n]     = (_Float16)((0.5f * LOG2E) * q[n]);   // scale+log2e
        Qh16[t * 16 + 8 + n] = (_Float16)0;
        Kh16[t * 16 + n]     = (_Float16)((n >= 4) ? -k[n] : k[n]); // fold metric c0
        Kh16[t * 16 + 8 + n] = (_Float16)0;
        Vt16[((size_t)b * 16 + n) * S_ + si] = (_Float16)v[n];
    }
    Vt16[((size_t)b * 16 + 8) * S_ + si] = (_Float16)1.0f;          // ones row
#pragma unroll
    for (int n = 9; n < 16; ++n)
        Vt16[((size_t)b * 16 + n) * S_ + si] = (_Float16)0;
}

// ---- Pass 2: fused flash attention, one 16-query tile per wave -------------
// S^T = K~ x Q^T (head dim 8; K-slots 8..31 of B are zero so the matching A
// slots are dont-care -> A operands are straight v16h loads, no masking).
// O^T += V^T x P^T: score-tile f32 C-layout == P^T f16 B-operand layout ->
// only per-lane sub+v_exp+cvt between matmuls; V row 8 (ones) makes the PV
// WMMA also produce the softmax denominator. 64 keys/iter, two accumulators.
__global__ __launch_bounds__(32 * WAVES) void clifford_attn_kernel(
    const _Float16* __restrict__ Qh16, const _Float16* __restrict__ Kh16,
    const _Float16* __restrict__ Vt16,
    const float* __restrict__ wo, const float* __restrict__ bo,
    float* __restrict__ out)
{
    const int lane  = threadIdx.x & 31;
    const int wave  = threadIdx.x >> 5;
    const int b     = blockIdx.x;
    const int qbase = (blockIdx.y * WAVES + wave) * 16;
    const bool lo   = lane < 16;
    const int  l16  = lane & 15;
    const int  hoff = lo ? 0 : 8;

    const v8h z8 = {};
    const v8f zc = {};

    // B operand for scores: Q^T, loop invariant. K-slots 8..31 = 0.
    v8h qr = *(const v8h*)(Qh16 + ((size_t)b * S_ + qbase + l16) * 16 + hoff);
    v16h bQ = __builtin_shufflevector(qr, z8, 0, 1, 2, 3, 4, 5, 6, 7,
                                      8, 9, 10, 11, 12, 13, 14, 15);

    const _Float16* Kbase = Kh16 + (size_t)b * S_ * 16;
    // PV A-operand rows: 0..7 = V dims, 8 = ones (denominator), 9..15 = 0.
    const _Float16* Vbase = Vt16 + (size_t)b * 16 * S_ + (size_t)l16 * S_;

    v8f oacc0 = {}, oacc1 = {};
    float m = -3.0e38f;

    for (int kc = 0; kc < S_; kc += 64) {
        // ---- 4 x 16-key score tiles: A = full 32B K~ rows, same address for
        // lanes l and l+16 (hi-lane slots 8..15 are dont-care, killed by bQ zeros) --
        const _Float16* kp = Kbase + (size_t)(kc + l16) * 16;
        v16h aK0 = *(const v16h*)(kp);
        v16h aK1 = *(const v16h*)(kp + 16 * 16);
        v16h aK2 = *(const v16h*)(kp + 32 * 16);
        v16h aK3 = *(const v16h*)(kp + 48 * 16);

        v8f s0 = __builtin_amdgcn_wmma_f32_16x16x32_f16(false, aK0, false, bQ,
                                                        (short)0, zc, false, false);
        v8f s1 = __builtin_amdgcn_wmma_f32_16x16x32_f16(false, aK1, false, bQ,
                                                        (short)0, zc, false, false);
        v8f s2 = __builtin_amdgcn_wmma_f32_16x16x32_f16(false, aK2, false, bQ,
                                                        (short)0, zc, false, false);
        v8f s3 = __builtin_amdgcn_wmma_f32_16x16x32_f16(false, aK3, false, bQ,
                                                        (short)0, zc, false, false);

        // ---- online softmax (log2 space); each lane owns 32 of the 64 chunk
        // keys for its query, the xor-16 partner owns the other 32 ----
        float lm = s0[0];
#pragma unroll
        for (int g = 1; g < 8; ++g) lm = fmaxf(lm, s0[g]);
#pragma unroll
        for (int g = 0; g < 8; ++g) lm = fmaxf(lm, s1[g]);
#pragma unroll
        for (int g = 0; g < 8; ++g) lm = fmaxf(lm, s2[g]);
#pragma unroll
        for (int g = 0; g < 8; ++g) lm = fmaxf(lm, s3[g]);
        lm = fmaxf(lm, __shfl_xor(lm, 16, 32));
        float nm   = fmaxf(m, lm);
        // args are <= 0; hardware v_exp_f32 flushes deep-negative to ~0,
        // which is exactly right for softmax tails (no libm range guard).
        float corr = __builtin_amdgcn_exp2f(m - nm);
        m = nm;

        v16h bP0, bP1;
#pragma unroll
        for (int g = 0; g < 8; ++g) {
            bP0[g]     = (_Float16)__builtin_amdgcn_exp2f(s0[g] - nm); // kc+0..7 / 8..15
            bP0[8 + g] = (_Float16)__builtin_amdgcn_exp2f(s1[g] - nm); // kc+16..23 / 24..31
            bP1[g]     = (_Float16)__builtin_amdgcn_exp2f(s2[g] - nm); // kc+32..39 / 40..47
            bP1[8 + g] = (_Float16)__builtin_amdgcn_exp2f(s3[g] - nm); // kc+48..55 / 56..63
        }
#pragma unroll
        for (int g = 0; g < 8; ++g) { oacc0[g] *= corr; oacc1[g] *= corr; }

        // ---- O^T += V^T x P^T (row 8 of V = ones -> denominator) ----
        const _Float16* vp = Vbase + kc + hoff;
        v8h v0 = *(const v8h*)(vp);
        v8h v1 = *(const v8h*)(vp + 16);
        v8h v2 = *(const v8h*)(vp + 32);
        v8h v3 = *(const v8h*)(vp + 48);
        v16h aV0 = __builtin_shufflevector(v0, v1, 0,1,2,3,4,5,6,7,8,9,10,11,12,13,14,15);
        v16h aV1 = __builtin_shufflevector(v2, v3, 0,1,2,3,4,5,6,7,8,9,10,11,12,13,14,15);

        oacc0 = __builtin_amdgcn_wmma_f32_16x16x32_f16(false, aV0, false, bP0,
                                                       (short)0, oacc0, false, false);
        oacc1 = __builtin_amdgcn_wmma_f32_16x16x32_f16(false, aV1, false, bP1,
                                                       (short)0, oacc1, false, false);
    }

    // Softmax denominator sits in output row 8 = hi-lane VGPR0.
    float sumv = oacc0[0] + oacc1[0];
    float tot  = __shfl_xor(sumv, 16, 32);   // lo lanes receive the row-8 sum

    if (lo) {
        int q = qbase + l16;
        float inv = 1.0f / tot;
        float orow[8];
#pragma unroll
        for (int g = 0; g < 8; ++g) orow[g] = (oacc0[g] + oacc1[g]) * inv;

        // final Clifford projection: y = orow . Wo + bo
        float y[8];
#pragma unroll
        for (int n = 0; n < 8; ++n) y[n] = bo[n];
#pragma unroll
        for (int i = 0; i < 8; ++i) {
            float woi = wo[i];
#pragma unroll
            for (int j = 0; j < 8; ++j) {
                int kk; float s;
                geo(i, j, kk, s);
                y[kk] += woi * s * orow[j];
            }
        }
        float* op = out + ((size_t)b * S_ + q) * 8;
#pragma unroll
        for (int n = 0; n < 8; ++n) op[n] = y[n];
    }
}

extern "C" void kernel_launch(void* const* d_in, const int* in_sizes, int n_in,
                              void* d_out, int out_size, void* d_ws, size_t ws_size,
                              hipStream_t stream) {
    const float* x  = (const float*)d_in[0];
    const float* wq = (const float*)d_in[1];
    const float* bq = (const float*)d_in[2];
    const float* wk = (const float*)d_in[3];
    const float* bk = (const float*)d_in[4];
    const float* wv = (const float*)d_in[5];
    const float* bv = (const float*)d_in[6];
    const float* wo = (const float*)d_in[7];
    const float* bo = (const float*)d_in[8];
    float* out = (float*)d_out;

    _Float16* Qh16 = (_Float16*)d_ws;
    _Float16* Kh16 = Qh16 + (size_t)B_ * S_ * 16;
    _Float16* Vt16 = Kh16 + (size_t)B_ * S_ * 16;

    int tokens = B_ * S_;
    clifford_qkv_kernel<<<(tokens + 255) / 256, 256, 0, stream>>>(
        x, wq, bq, wk, bk, wv, bv, Qh16, Kh16, Vt16);

    dim3 grid(B_, S_ / (16 * WAVES));
    clifford_attn_kernel<<<grid, 32 * WAVES, 0, stream>>>(
        Qh16, Kh16, Vt16, wo, bo, out);
}
